// TriRank_15453292331473
// MI455X (gfx1250) — matrix operand order repaired and checked by reading, not compiled, not run on hardware
//
#include <hip/hip_runtime.h>
#include <hip/hip_bf16.h>

typedef __bf16 bf16_t;
typedef __attribute__((ext_vector_type(16))) __bf16 bf16x16;
typedef __attribute__((ext_vector_type(8)))  float  f32x8;

#define U_NUM 2048
#define I_NUM 4096
#define A_NUM 1024

// ---------------------------------------------------------------------------
// Preprocessing kernels
// ---------------------------------------------------------------------------

__global__ void rowsum_kernel(const float* __restrict__ W, int M, int N,
                              float* __restrict__ out) {
  __shared__ float red[256];
  const int m = blockIdx.x;
  float s = 0.f;
  for (int n = threadIdx.x; n < N; n += blockDim.x) s += W[(size_t)m * N + n];
  red[threadIdx.x] = s;
  __syncthreads();
  for (int off = 128; off > 0; off >>= 1) {
    if ((int)threadIdx.x < off) red[threadIdx.x] += red[threadIdx.x + off];
    __syncthreads();
  }
  if (threadIdx.x == 0) out[m] = red[0];
}

__global__ void colsum_kernel(const float* __restrict__ W, int M, int N,
                              float* __restrict__ out) {
  const int n = blockIdx.x * blockDim.x + threadIdx.x;
  if (n >= N) return;
  float s = 0.f;
  for (int m = 0; m < M; ++m) s += W[(size_t)m * N + n];
  out[n] = s;
}

// Row sums of the normalized matrix S = W / (row*col + eps)
__global__ void snorm_rowsum_kernel(const float* __restrict__ W,
                                    const float* __restrict__ row,
                                    const float* __restrict__ col,
                                    int M, int N, float* __restrict__ out) {
  __shared__ float red[256];
  const int m = blockIdx.x;
  const float rm = row[m];
  float s = 0.f;
  for (int n = threadIdx.x; n < N; n += blockDim.x)
    s += W[(size_t)m * N + n] / (rm * col[n] + 1e-7f);
  red[threadIdx.x] = s;
  __syncthreads();
  for (int off = 128; off > 0; off >>= 1) {
    if ((int)threadIdx.x < off) red[threadIdx.x] += red[threadIdx.x + off];
    __syncthreads();
  }
  if (threadIdx.x == 0) out[m] = red[0];
}

// S[m,n]=W/(row*col+eps) -> bf16, transpose -> bf16, optional T0T[n,m]=S/srow[m]
__global__ void build_kernel(const float* __restrict__ W,
                             const float* __restrict__ row,
                             const float* __restrict__ col,
                             const float* __restrict__ srow,
                             int M, int N,
                             bf16_t* __restrict__ S, bf16_t* __restrict__ ST,
                             float* __restrict__ T0T) {
  const size_t idx = blockIdx.x * (size_t)blockDim.x + threadIdx.x;
  if (idx >= (size_t)M * N) return;
  const int m = (int)(idx / N);
  const int n = (int)(idx % N);
  const float s = W[idx] / (row[m] * col[n] + 1e-7f);
  S[idx] = (bf16_t)s;
  ST[(size_t)n * M + m] = (bf16_t)s;
  if (T0T) T0T[(size_t)n * M + m] = s / srow[m];
}

__global__ void f2bf_kernel(const float* __restrict__ x, bf16_t* __restrict__ y,
                            size_t n) {
  const size_t i = blockIdx.x * (size_t)blockDim.x + threadIdx.x;
  if (i < n) y[i] = (bf16_t)x[i];
}

// out (C x R) = X^T where X is (R x C); R,C multiples of 32
__global__ void transpose_kernel(const float* __restrict__ X, int R, int C,
                                 float* __restrict__ out) {
  __shared__ float tile[32][33];
  const int c0 = blockIdx.x * 32;
  const int r0 = blockIdx.y * 32;
  const int x = threadIdx.x;
  const int y = threadIdx.y;  // block (32,8)
#pragma unroll
  for (int j = 0; j < 32; j += 8)
    tile[y + j][x] = X[(size_t)(r0 + y + j) * C + c0 + x];
  __syncthreads();
#pragma unroll
  for (int j = 0; j < 32; j += 8)
    out[(size_t)(c0 + y + j) * R + r0 + x] = tile[x][y + j];
}

// ---------------------------------------------------------------------------
// Fused double-GEMM stage:  C = s2 * ( (s1/s2) * A1@B1 + A2@B2 ) + z
// ZIDENT==1: z = zscale on the diagonal;  ZIDENT==0: z = zscale * Z[m,n]
// A row-major MxK bf16, B row-major KxN bf16. Writes bf16 (+ f32 if WRITEF).
// M,N multiples of 128; K1,K2 multiples of 32.
// LDS tiles double-buffered, row stride padded to 40 elems (80B = 20 dwords)
// so both the transposed B staging stores and the per-lane fragment reads are
// LDS-bank-conflict free, while every 16B access stays 16B-aligned.
// ---------------------------------------------------------------------------
constexpr int BM = 128, BN = 128, BK = 32, LDT = 40;

struct TileRegs { uint4 a0, a1, b0, b1; };

__device__ __forceinline__ TileRegs fetch_tile(const bf16_t* __restrict__ Ag,
                                               const bf16_t* __restrict__ Bg,
                                               int K, int N, int bm, int bn,
                                               int k0, int tid) {
  TileRegs t;
  // A tile: 128x32, 512 16B chunks, rows coalesced.
  int chunk = tid;
  int r = chunk >> 2, c = (chunk & 3) * 8;
  t.a0 = *reinterpret_cast<const uint4*>(Ag + (bm + r) * K + k0 + c);
  chunk = tid + 256;
  r = chunk >> 2; c = (chunk & 3) * 8;
  t.a1 = *reinterpret_cast<const uint4*>(Ag + (bm + r) * K + k0 + c);
  // B tile: 32x128, rows coalesced.
  chunk = tid;
  int kr = chunk >> 4; c = (chunk & 15) * 8;
  t.b0 = *reinterpret_cast<const uint4*>(Bg + (k0 + kr) * N + bn + c);
  chunk = tid + 256;
  kr = chunk >> 4; c = (chunk & 15) * 8;
  t.b1 = *reinterpret_cast<const uint4*>(Bg + (k0 + kr) * N + bn + c);
  return t;
}

__device__ __forceinline__ void store_tile(bf16_t* __restrict__ As,
                                           bf16_t* __restrict__ Bs,
                                           const TileRegs& t, int tid) {
  int chunk = tid;
  int r = chunk >> 2, c = (chunk & 3) * 8;
  *reinterpret_cast<uint4*>(&As[r * LDT + c]) = t.a0;
  chunk = tid + 256;
  r = chunk >> 2; c = (chunk & 3) * 8;
  *reinterpret_cast<uint4*>(&As[r * LDT + c]) = t.a1;

  chunk = tid;
  int kr = chunk >> 4; c = (chunk & 15) * 8;
  const bf16_t* p0 = reinterpret_cast<const bf16_t*>(&t.b0);
#pragma unroll
  for (int e = 0; e < 8; ++e) Bs[(c + e) * LDT + kr] = p0[e];
  chunk = tid + 256;
  kr = chunk >> 4; c = (chunk & 15) * 8;
  const bf16_t* p1 = reinterpret_cast<const bf16_t*>(&t.b1);
#pragma unroll
  for (int e = 0; e < 8; ++e) Bs[(c + e) * LDT + kr] = p1[e];
}

template <int ZIDENT, bool WRITEF>
__global__ __launch_bounds__(256) void stage_kernel(
    const bf16_t* __restrict__ A1, const bf16_t* __restrict__ B1, int K1,
    const bf16_t* __restrict__ A2, const bf16_t* __restrict__ B2, int K2,
    float s1, float s2,
    const float* __restrict__ Z, float zscale,
    float* __restrict__ Cf, bf16_t* __restrict__ Cb, int M, int N) {
  __shared__ alignas(16) bf16_t As[2][BM * LDT];
  __shared__ alignas(16) bf16_t Bs[2][BN * LDT];  // transposed: [n][k]

  const int tid = threadIdx.x;
  const int lane = tid & 31;
  const int wid = tid >> 5;        // 8 waves: 2 x 4
  const int wm = (wid >> 2) * 64;  // wave M offset in block tile
  const int wn = (wid & 3) * 32;   // wave N offset
  const int lh = lane >> 4;        // lane half (0/1)
  const int lm = lane & 15;

  const int bm = blockIdx.y * BM;
  const int bn = blockIdx.x * BN;

  f32x8 acc[4][2] = {};  // 64 f32 accumulators per lane

  const int T1 = K1 / BK;
  const int T = T1 + K2 / BK;

  // Prologue: stage tile 0.
  {
    TileRegs t0 = fetch_tile(A1, B1, K1, N, bm, bn, 0, tid);
    store_tile(As[0], Bs[0], t0, tid);
    __syncthreads();
  }

  int pb = 0;
  for (int t = 0; t < T; ++t) {
    const bool more = (t + 1 < T);
    TileRegs nxt;
    if (more) {
      const int tn = t + 1;
      if (tn < T1) nxt = fetch_tile(A1, B1, K1, N, bm, bn, tn * BK, tid);
      else         nxt = fetch_tile(A2, B2, K2, N, bm, bn, (tn - T1) * BK, tid);
    }

    // Build fragments in contiguous k order (k = 16*lh + e), same for A and B.
    const bf16_t* __restrict__ Ab = As[pb];
    const bf16_t* __restrict__ Bb = Bs[pb];
    bf16x16 afrag[4], bfrag[2];
#pragma unroll
    for (int mi = 0; mi < 4; ++mi) {
      const bf16_t* rowp = &Ab[(wm + mi * 16 + lm) * LDT];
      // lanes 0-15: K 0..15 ; lanes 16-31: K 16..31
#pragma unroll
      for (int e = 0; e < 8; ++e) {
        afrag[mi][e]     = rowp[lh * 16 + e];
        afrag[mi][8 + e] = rowp[lh * 16 + 8 + e];
      }
    }
#pragma unroll
    for (int ni = 0; ni < 2; ++ni) {
      // lane holds column n = lm, K = lh*16 .. lh*16+15 (contiguous in Bs)
      const bf16_t* colp = &Bb[(wn + ni * 16 + lm) * LDT + lh * 16];
#pragma unroll
      for (int e = 0; e < 16; ++e) bfrag[ni][e] = colp[e];
    }

#pragma unroll
    for (int mi = 0; mi < 4; ++mi)
#pragma unroll
      for (int ni = 0; ni < 2; ++ni)
        acc[mi][ni] = __builtin_amdgcn_wmma_f32_16x16x32_bf16(
            false, afrag[mi], false, bfrag[ni], (short)0, acc[mi][ni], false,
            false);

    // Segment boundary: rescale so acc = (s1/s2)*P1 + P2; epilogue applies s2.
    if (t == T1 - 1) {
      const float sc = s1 / s2;
#pragma unroll
      for (int i = 0; i < 4; ++i)
#pragma unroll
        for (int j = 0; j < 2; ++j)
#pragma unroll
          for (int r = 0; r < 8; ++r) acc[i][j][r] *= sc;
    }

    if (more) {
      store_tile(As[pb ^ 1], Bs[pb ^ 1], nxt, tid);
      __syncthreads();
      pb ^= 1;
    }
  }

  // Epilogue: C layout — VGPR r: lanes 0-15 -> M=r, lanes 16-31 -> M=8+r.
  (void)M;
#pragma unroll
  for (int mi = 0; mi < 4; ++mi) {
#pragma unroll
    for (int r = 0; r < 8; ++r) {
      const int m = bm + wm + mi * 16 + lh * 8 + r;
      const int rowoff = m * N;  // < 2^31, 32-bit math
#pragma unroll
      for (int ni = 0; ni < 2; ++ni) {
        const int n = bn + wn + ni * 16 + lm;
        float v = acc[mi][ni][r] * s2;
        if (ZIDENT) {
          v += (m == n) ? zscale : 0.f;
        } else {
          v = fmaf(zscale, Z[rowoff + n], v);
        }
        if (WRITEF) Cf[rowoff + n] = v;
        Cb[rowoff + n] = (bf16_t)v;
      }
    }
  }
}

// ---------------------------------------------------------------------------
// Host-side orchestration
// ---------------------------------------------------------------------------
extern "C" void kernel_launch(void* const* d_in, const int* in_sizes, int n_in,
                              void* d_out, int out_size, void* d_ws,
                              size_t ws_size, hipStream_t stream) {
  (void)in_sizes; (void)n_in; (void)out_size; (void)ws_size;
  const int U = U_NUM, I = I_NUM, A = A_NUM;

  const float* userItem   = (const float*)d_in[0];  // U x I
  const float* userAspect = (const float*)d_in[1];  // U x A
  const float* itemAspect = (const float*)d_in[2];  // I x A
  // d_in[3] (userVec0) is never read by the recurrence.
  const float* itemVec0   = (const float*)d_in[4];  // I x U
  const float* aspectVec0 = (const float*)d_in[5];  // A x U

  size_t off = 0;
  auto alloc = [&](size_t bytes) -> char* {
    char* p = (char*)d_ws + off;
    off += (bytes + 255) & ~(size_t)255;
    return p;
  };

  float* row_ui  = (float*)alloc((size_t)U * 4);
  float* col_ui  = (float*)alloc((size_t)I * 4);
  float* row_ua  = (float*)alloc((size_t)U * 4);
  float* col_ua  = (float*)alloc((size_t)A * 4);
  float* row_ia  = (float*)alloc((size_t)I * 4);
  float* col_ia  = (float*)alloc((size_t)A * 4);
  float* srow_ui = (float*)alloc((size_t)U * 4);
  float* srow_ua = (float*)alloc((size_t)U * 4);

  bf16_t* S_ui = (bf16_t*)alloc((size_t)U * I * 2);
  bf16_t* S_iu = (bf16_t*)alloc((size_t)U * I * 2);
  bf16_t* S_ua = (bf16_t*)alloc((size_t)U * A * 2);
  bf16_t* S_au = (bf16_t*)alloc((size_t)U * A * 2);
  bf16_t* S_ia = (bf16_t*)alloc((size_t)I * A * 2);
  bf16_t* S_ai = (bf16_t*)alloc((size_t)I * A * 2);

  float* I0 = (float*)alloc((size_t)I * U * 4);  // (I x U)
  float* A0 = (float*)alloc((size_t)A * U * 4);  // (A x U)

  bf16_t* uv_b = (bf16_t*)alloc((size_t)U * U * 2);
  float*  iv_f = (float*) alloc((size_t)I * U * 4);
  bf16_t* iv_b = (bf16_t*)alloc((size_t)I * U * 2);
  float*  av_f = (float*) alloc((size_t)A * U * 4);
  bf16_t* av_b = (bf16_t*)alloc((size_t)A * U * 2);

  // --- sums ---
  rowsum_kernel<<<U, 256, 0, stream>>>(userItem, U, I, row_ui);
  colsum_kernel<<<I / 256, 256, 0, stream>>>(userItem, U, I, col_ui);
  rowsum_kernel<<<U, 256, 0, stream>>>(userAspect, U, A, row_ua);
  colsum_kernel<<<A / 256, 256, 0, stream>>>(userAspect, U, A, col_ua);
  rowsum_kernel<<<I, 256, 0, stream>>>(itemAspect, I, A, row_ia);
  colsum_kernel<<<A / 256, 256, 0, stream>>>(itemAspect, I, A, col_ia);

  snorm_rowsum_kernel<<<U, 256, 0, stream>>>(userItem, row_ui, col_ui, U, I, srow_ui);
  snorm_rowsum_kernel<<<U, 256, 0, stream>>>(userAspect, row_ua, col_ua, U, A, srow_ua);

  // --- normalized matrices (+ transposes, + I0/A0) ---
  {
    size_t n = (size_t)U * I;
    build_kernel<<<(unsigned)((n + 255) / 256), 256, 0, stream>>>(
        userItem, row_ui, col_ui, srow_ui, U, I, S_ui, S_iu, I0);
    n = (size_t)U * A;
    build_kernel<<<(unsigned)((n + 255) / 256), 256, 0, stream>>>(
        userAspect, row_ua, col_ua, srow_ua, U, A, S_ua, S_au, A0);
    n = (size_t)I * A;
    build_kernel<<<(unsigned)((n + 255) / 256), 256, 0, stream>>>(
        itemAspect, row_ia, col_ia, nullptr, I, A, S_ia, S_ai, nullptr);
  }

  // --- initial iterates (bf16 copies) ---
  {
    size_t n = (size_t)I * U;
    f2bf_kernel<<<(unsigned)((n + 255) / 256), 256, 0, stream>>>(itemVec0, iv_b, n);
    n = (size_t)A * U;
    f2bf_kernel<<<(unsigned)((n + 255) / 256), 256, 0, stream>>>(aspectVec0, av_b, n);
  }

  const float third = 1.0f / 3.0f;  // ALPHA=BETA=GAMMA=*0=1 -> ca=cb=cc=3
  const dim3 blk(256);
  const dim3 grid_u(U / 128, U / 128);
  const dim3 grid_i(U / 128, I / 128);
  const dim3 grid_a(U / 128, A / 128);

  for (int it = 0; it < 10; ++it) {
    // uv = (S_ui@iv + S_ua@av + Ident)/3   (f32 copy not needed)
    stage_kernel<1, false><<<grid_u, blk, 0, stream>>>(
        S_ui, iv_b, I, S_ua, av_b, A, third, third,
        nullptr, third, nullptr, uv_b, U, U);
    // iv = (S_iu@uv + S_ia@av + I0)/3      (uses OLD av)
    stage_kernel<0, true><<<grid_i, blk, 0, stream>>>(
        S_iu, uv_b, U, S_ia, av_b, A, third, third,
        I0, third, iv_f, iv_b, I, U);
    // av = (S_au@uv + S_ai@iv + A0)/3      (uses NEW uv, NEW iv)
    stage_kernel<0, true><<<grid_a, blk, 0, stream>>>(
        S_au, uv_b, U, S_ai, iv_b, I, third, third,
        A0, third, av_f, av_b, A, U);
  }

  // --- outputs: iv.T (U x I) then av.T (U x A), concatenated flat ---
  float* out = (float*)d_out;
  {
    dim3 tb(32, 8);
    dim3 g1(U / 32, I / 32);  // X = iv_f (I x U) -> out (U x I)
    transpose_kernel<<<g1, tb, 0, stream>>>(iv_f, I, U, out);
    dim3 g2(U / 32, A / 32);  // X = av_f (A x U) -> out (U x A)
    transpose_kernel<<<g2, tb, 0, stream>>>(av_f, A, U, out + (size_t)U * I);
  }
}
